// ExpertAttention_64235530879009
// MI455X (gfx1250) — matrix operand order repaired and checked by reading, not compiled, not run on hardware
//
#include <hip/hip_runtime.h>
#include <hip/hip_bf16.h>

#define DEV __device__ __forceinline__

typedef _Float16 f16;
typedef _Float16 v16h __attribute__((ext_vector_type(16)));
typedef float    v8f  __attribute__((ext_vector_type(8)));
typedef int      i32x4 __attribute__((ext_vector_type(4)));

static constexpr int Bz = 32, Sq = 512, Dm = 768, Hh = 12, DHe = 64;

// ---------------------------------------------------------------------------
// CDNA5 async global->LDS copy (ASYNCcnt-tracked), guarded so the file still
// compiles (sync fallback) if the toolchain lacks the builtins.
// ---------------------------------------------------------------------------
#if defined(__AMDGCN__) &&                                            \
    __has_builtin(__builtin_amdgcn_global_load_async_to_lds_b128) &&  \
    __has_builtin(__builtin_amdgcn_s_wait_asynccnt)
#define ASYNC_LDS 1
#else
#define ASYNC_LDS 0
#endif

#if ASYNC_LDS
DEV void async_b128(const f16* g, f16* l) {
  __builtin_amdgcn_global_load_async_to_lds_b128(
      (__attribute__((address_space(1))) i32x4*)g,
      (__attribute__((address_space(3))) i32x4*)l, 0, 0);
}
#define ASYNC_WAIT() __builtin_amdgcn_s_wait_asynccnt(0)
#else
#define ASYNC_WAIT() ((void)0)
#endif

// ---------------------------------------------------------------------------
// WMMA helpers (CDNA5 v_wmma_f32_16x16x32_f16, wave32)
// ---------------------------------------------------------------------------
DEV v8f wmma16(v16h a, v16h b, v8f c) {
  return __builtin_amdgcn_wmma_f32_16x16x32_f16(
      /*neg_a=*/false, a, /*neg_b=*/false, b,
      /*c_mod=*/(short)0, c, /*reuse_a=*/false, /*reuse_b=*/false);
}

union FragU { v16h v; unsigned u[8]; };

// A fragment, 16x32 f16, source tile row-major [row][k], row stride `ld`.
// ISA: lane<16 row=lane, K=0..7 / 16..23; lane>=16 same row, K=8..15 / 24..31.
DEV v16h load_frag_A(const f16* base, int ld, int row, int lane) {
  FragU f;
  const f16* p = base + row * ld + ((lane >= 16) ? 8 : 0);
#pragma unroll
  for (int jj = 0; jj < 8; ++jj) {
    int k = 2 * jj + ((jj >= 4) ? 8 : 0);
    f.u[jj] = *(const unsigned*)(p + k);
  }
  return f.v;
}

// B fragment, 32x16 f16, source stored TRANSPOSED [n][k], stride `ld`.
// ISA: lane<16 col=lane K=0..7 / 16..23; lane>=16 col=lane-16 K=8..15 / 24..31.
DEV v16h load_frag_BT(const f16* base, int ld, int col, int lane) {
  FragU f;
  const f16* p = base + col * ld + ((lane >= 16) ? 8 : 0);
#pragma unroll
  for (int jj = 0; jj < 8; ++jj) f.u[jj] = *(const unsigned*)(p + 2 * jj + ((jj >= 4) ? 8 : 0));
  return f.v;
}

// ---------------------------------------------------------------------------
// Stage 1: routing (mean over S, argmin distance to E=2 centers)
// ---------------------------------------------------------------------------
__global__ __launch_bounds__(256) void route_kernel(
    const float* __restrict__ X, const float* __restrict__ centers,
    int* __restrict__ assign) {
  const int b = blockIdx.x, tid = threadIdx.x;
  const float* xb = X + (size_t)b * Sq * Dm;
  float a0 = 0.f, a1 = 0.f, a2 = 0.f;
  for (int s = 0; s < Sq; ++s) {
    const float* row = xb + s * Dm;
    a0 += row[tid]; a1 += row[tid + 256]; a2 += row[tid + 512];
  }
  const float inv = 1.0f / (float)Sq;
  a0 *= inv; a1 *= inv; a2 *= inv;
  __shared__ float red[256];
  float d2[2];
#pragma unroll
  for (int e = 0; e < 2; ++e) {
    const float* c = centers + e * Dm;
    float p0 = a0 - c[tid], p1 = a1 - c[tid + 256], p2 = a2 - c[tid + 512];
    red[tid] = p0 * p0 + p1 * p1 + p2 * p2;
    __syncthreads();
    for (int ofs = 128; ofs > 0; ofs >>= 1) {
      if (tid < ofs) red[tid] += red[tid + ofs];
      __syncthreads();
    }
    d2[e] = red[0];
    __syncthreads();
  }
  if (tid == 0) assign[b] = (d2[1] < d2[0]) ? 1 : 0;
}

// ---------------------------------------------------------------------------
// Stage 2: f32 -> f16 cast
// ---------------------------------------------------------------------------
__global__ __launch_bounds__(256) void cast_f16_kernel(
    const float* __restrict__ src, f16* __restrict__ dst, int n) {
  int i = blockIdx.x * 256 + threadIdx.x;
  if (i < n) dst[i] = (f16)src[i];
}

// ---------------------------------------------------------------------------
// Stage 3/5: C[b] = X[b](512x768) @ W[assign[b]](768x768) + bias[assign[b]]
// 128x128 block tile, BK=32, 8 waves (4x2), double-buffered LDS,
// async global->LDS for the A tile, register-transpose path for B.
// ---------------------------------------------------------------------------
template <bool OUT_F16>
__global__ __launch_bounds__(256) void gemm_bias_kernel(
    const f16* __restrict__ X,      // [B, 512, 768] f16
    const f16* __restrict__ W,      // [E, 768, 768] f16
    const float* __restrict__ bias, // [E, 768] f32
    const int* __restrict__ assign,
    f16* __restrict__ outH, float* __restrict__ outF) {
  constexpr int BM = 128, BN = 128, BK = 32, LDA = BK + 8, LDB = BK + 8;
  constexpr int NT = Dm / BK;
  __shared__ f16 sA[2][BM * LDA];   // [m][k]
  __shared__ f16 sBT[2][BN * LDB];  // [n][k] (transposed)

  const int b = blockIdx.z;
  const int e = assign[b];
  const int m0 = blockIdx.x * BM, n0 = blockIdx.y * BN;
  const f16* Xb = X + (size_t)b * Sq * Dm;
  const f16* We = W + (size_t)e * Dm * Dm;
  const int tid = threadIdx.x, lane = tid & 31, wave = tid >> 5;
  const int wm = wave >> 1, wn = wave & 1;   // 4x2 wave grid
  const int lrow = lane & 15;

  auto load_tile = [&](int buf, int k0) {
#if ASYNC_LDS
    // A tile: 128x32 f16 = 512 b128 chunks, 2 per thread (async DMA to LDS)
#pragma unroll
    for (int i = 0; i < 2; ++i) {
      int p = (tid + i * 256) * 8;
      int r = p >> 5, c = p & 31;
      async_b128(Xb + (size_t)(m0 + r) * Dm + k0 + c, &sA[buf][r * LDA + c]);
    }
#else
#pragma unroll
    for (int i = 0; i < 8; ++i) {
      int p = (tid + i * 256) * 2;
      int r = p >> 5, c = p & 31;
      unsigned v = *(const unsigned*)(Xb + (size_t)(m0 + r) * Dm + k0 + c);
      *(unsigned*)(&sA[buf][r * LDA + c]) = v;
    }
#endif
    // B tile: 32x128, read row-major dwords, scatter transposed halves
#pragma unroll
    for (int i = 0; i < 8; ++i) {
      int p = (tid + i * 256) * 2;
      int kr = p >> 7, nc = p & 127;
      unsigned v = *(const unsigned*)(We + (size_t)(k0 + kr) * Dm + n0 + nc);
      union { unsigned u; f16 h[2]; } cv; cv.u = v;
      sBT[buf][nc * LDB + kr] = cv.h[0];
      sBT[buf][(nc + 1) * LDB + kr] = cv.h[1];
    }
  };

  v8f acc[2][4];
#pragma unroll
  for (int i = 0; i < 2; ++i)
#pragma unroll
    for (int j = 0; j < 4; ++j) acc[i][j] = {};

  load_tile(0, 0);
  for (int kt = 0; kt < NT; ++kt) {
    const int cur = kt & 1;
    ASYNC_WAIT();      // this wave's pending async copies done
    __syncthreads();   // all waves' tile data visible; prev readers drained
    if (kt + 1 < NT) load_tile(1 - cur, (kt + 1) * BK);

    v16h af[2], bf[4];
#pragma unroll
    for (int i = 0; i < 2; ++i)
      af[i] = load_frag_A(sA[cur], LDA, wm * 32 + i * 16 + lrow, lane);
#pragma unroll
    for (int j = 0; j < 4; ++j)
      bf[j] = load_frag_BT(sBT[cur], LDB, wn * 64 + j * 16 + lrow, lane);
#pragma unroll
    for (int i = 0; i < 2; ++i)
#pragma unroll
      for (int j = 0; j < 4; ++j) acc[i][j] = wmma16(af[i], bf[j], acc[i][j]);
  }

  const float* be = bias + (size_t)e * Dm;
#pragma unroll
  for (int i = 0; i < 2; ++i) {
    int mrow = m0 + wm * 32 + i * 16 + ((lane >= 16) ? 8 : 0);
#pragma unroll
    for (int j = 0; j < 4; ++j) {
      int gn = n0 + wn * 64 + j * 16 + lrow;
      float bb = be[gn];
#pragma unroll
      for (int r = 0; r < 8; ++r) {
        float v = acc[i][j][r] + bb;
        size_t oi = ((size_t)b * Sq + mrow + r) * Dm + gn;
        if constexpr (OUT_F16) outH[oi] = (f16)v; else outF[oi] = v;
      }
    }
  }
}

// ---------------------------------------------------------------------------
// Stage 4: attention per (b, h, 32-row q tile). Full-row softmax in LDS.
// Double-buffered K and V tiles so async DMA / global loads overlap WMMA.
// ---------------------------------------------------------------------------
__global__ __launch_bounds__(256) void attn_kernel(
    const f16* __restrict__ Q, const f16* __restrict__ K,
    const f16* __restrict__ V, const float* __restrict__ mask,
    f16* __restrict__ Ctx) {
  constexpr int QT = 32, KT = 64, NKT = Sq / KT;
  constexpr int LQ = DHe + 8, LK = DHe + 8, LV = KT + 8, LS = Sq + 8;
  __shared__ f16   sQ[QT * LQ];        // [q][dh]
  __shared__ f16   sK[2][KT * LK];     // [kv][dh]  == B^T for scores
  __shared__ f16   sVT[2][DHe * LV];   // [dh][kv]  == B^T for context
  __shared__ float sS[QT * LS];        // scores, full 512 kv
  __shared__ f16   sP[QT * LS];        // probabilities (f16)
  __shared__ float sRed[QT * 8];
  __shared__ float sRow[QT];

  const int bh = blockIdx.x, b = bh / Hh, h = bh % Hh;
  const int q0 = blockIdx.y * QT;
  const int tid = threadIdx.x, lane = tid & 31, wave = tid >> 5;
  const int lrow = lane & 15;
  const size_t headOfs = (size_t)b * Sq * Dm + (size_t)h * DHe;

  auto load_k = [&](int buf, int kt) {
#if ASYNC_LDS
    int r = tid >> 3, c = (tid & 7) * 8;   // 256 b128 chunks, 1 per thread
    async_b128(K + headOfs + (size_t)(kt + r) * Dm + c, &sK[buf][r * LK + c]);
#else
#pragma unroll
    for (int i = 0; i < 8; ++i) {
      int p = (tid + i * 256) * 2;
      int r = p >> 6, c = p & 63;
      *(unsigned*)(&sK[buf][r * LK + c]) =
          *(const unsigned*)(K + headOfs + (size_t)(kt + r) * Dm + c);
    }
#endif
  };
  auto load_v = [&](int buf, int kt) {  // V tile transposed into LDS [dh][kv]
#pragma unroll
    for (int i = 0; i < 8; ++i) {
      int p = (tid + i * 256) * 2;
      int r = p >> 6, c = p & 63;
      unsigned v = *(const unsigned*)(V + headOfs + (size_t)(kt + r) * Dm + c);
      union { unsigned u; f16 hh[2]; } cv; cv.u = v;
      sVT[buf][c * LV + r] = cv.hh[0];
      sVT[buf][(c + 1) * LV + r] = cv.hh[1];
    }
  };

  // load Q tile (32x64)
#pragma unroll
  for (int i = 0; i < 4; ++i) {
    int p = (tid + i * 256) * 2;
    int r = p >> 6, c = p & 63;
    *(unsigned*)(&sQ[r * LQ + c]) =
        *(const unsigned*)(Q + headOfs + (size_t)(q0 + r) * Dm + c);
  }
  __syncthreads();

  // ---- scores: 8 waves = 2 (q) x 4 (kv) tiles of 16x16 per 64-kv block ----
  const int wm = wave & 1, wn = wave >> 1;
  v16h aq[2];  // loop-invariant Q fragments (kk = 0, 32)
#pragma unroll
  for (int kk = 0; kk < 2; ++kk)
    aq[kk] = load_frag_A(sQ + kk * 32, LQ, wm * 16 + lrow, lane);

  load_k(0, 0);
  for (int kti = 0; kti < NKT; ++kti) {
    const int kt = kti * KT, cur = kti & 1;
    ASYNC_WAIT();      // this wave's pending K-tile DMA done
    __syncthreads();   // tile visible; prev readers of other buffer drained
    if (kti + 1 < NKT) load_k(1 - cur, kt + KT);

    v8f acc = {};
#pragma unroll
    for (int kk = 0; kk < 2; ++kk) {
      v16h bfr = load_frag_BT(sK[cur] + kk * 32, LK, wn * 16 + lrow, lane);
      acc = wmma16(aq[kk], bfr, acc);
    }
    int col = kt + wn * 16 + lrow;
    int rb = wm * 16 + ((lane >= 16) ? 8 : 0);
#pragma unroll
    for (int r = 0; r < 8; ++r) sS[(rb + r) * LS + col] = acc[r] * 0.125f;
  }
  __syncthreads();  // all score stripes visible before softmax

  // ---- softmax over 512 cols: thread (row = tid>>3, seg = tid&7, 64 cols) --
  {
    const int r = tid >> 3, seg = tid & 7, c0 = seg * 64;
    const float* mrow = mask + (size_t)b * Sq;
    float mx = -3.0e38f;
    for (int c = 0; c < 64; ++c) {
      float madd = (1.0f - mrow[c0 + c]) * (-10000.0f);
      float v = sS[r * LS + c0 + c] + madd;
      sS[r * LS + c0 + c] = v;
      mx = fmaxf(mx, v);
    }
    sRed[r * 8 + seg] = mx;
    __syncthreads();
    if (seg == 0) {
      float m = sRed[r * 8];
#pragma unroll
      for (int j = 1; j < 8; ++j) m = fmaxf(m, sRed[r * 8 + j]);
      sRow[r] = m;
    }
    __syncthreads();
    float rowm = sRow[r], sum = 0.f;
    for (int c = 0; c < 64; ++c) {
      float pv = __expf(sS[r * LS + c0 + c] - rowm);
      sS[r * LS + c0 + c] = pv;
      sum += pv;
    }
    sRed[r * 8 + seg] = sum;
    __syncthreads();
    if (seg == 0) {
      float s = sRed[r * 8];
#pragma unroll
      for (int j = 1; j < 8; ++j) s += sRed[r * 8 + j];
      sRow[r] = s;
    }
    __syncthreads();
    float inv = 1.0f / sRow[r];
    for (int c = 0; c < 64; ++c)
      sP[r * LS + c0 + c] = (f16)(sS[r * LS + c0 + c] * inv);
    __syncthreads();
  }

  // ---- context: 8 waves = 2 (q) x 4 (dh) tiles, accumulate over all kv ----
  v8f octx = {};
  load_v(0, 0);
  for (int kti = 0; kti < NKT; ++kti) {
    const int kt = kti * KT, cur = kti & 1;
    __syncthreads();   // V tile (cur) visible; prev readers drained
    if (kti + 1 < NKT) load_v(1 - cur, kt + KT);
#pragma unroll
    for (int kk = 0; kk < KT; kk += 32) {
      v16h a = load_frag_A(sP + kt + kk, LS, wm * 16 + lrow, lane);
      v16h bv = load_frag_BT(sVT[cur] + kk, LV, wn * 16 + lrow, lane);
      octx = wmma16(a, bv, octx);
    }
  }
  int grow = q0 + wm * 16 + ((lane >= 16) ? 8 : 0);
  int gcol = (wave >> 1) * 16 + lrow;
#pragma unroll
  for (int r = 0; r < 8; ++r)
    Ctx[(size_t)b * Sq * Dm + (size_t)(grow + r) * Dm + h * DHe + gcol] =
        (f16)octx[r];
}

// ---------------------------------------------------------------------------
// Launcher
// ---------------------------------------------------------------------------
extern "C" void kernel_launch(void* const* d_in, const int* in_sizes, int n_in,
                              void* d_out, int out_size, void* d_ws,
                              size_t ws_size, hipStream_t stream) {
  (void)in_sizes; (void)n_in; (void)out_size; (void)ws_size;
  const float* hidden  = (const float*)d_in[0];
  const float* amask   = (const float*)d_in[1];
  const float* centers = (const float*)d_in[2];
  const float* Wq = (const float*)d_in[3];
  const float* bq = (const float*)d_in[4];
  const float* Wk = (const float*)d_in[5];
  const float* bk = (const float*)d_in[6];
  const float* Wv = (const float*)d_in[7];
  const float* bv = (const float*)d_in[8];
  const float* Wo = (const float*)d_in[9];
  const float* bo = (const float*)d_in[10];
  float* out = (float*)d_out;

  const size_t szX = (size_t)Bz * Sq * Dm * sizeof(f16);   // 25.17 MB
  const size_t szW = (size_t)2 * Dm * Dm * sizeof(f16);    // 2.36 MB
  char* ws = (char*)d_ws;
  int* assign = (int*)ws;
  f16* Xh  = (f16*)(ws + 256);
  f16* Wqh = (f16*)(ws + 256 + szX);
  f16* Wkh = (f16*)(ws + 256 + szX + szW);
  f16* Wvh = (f16*)(ws + 256 + szX + 2 * szW);
  f16* Woh = (f16*)(ws + 256 + szX + 3 * szW);
  f16* Qh  = (f16*)(ws + 256 + szX + 4 * szW);
  f16* Kh  = (f16*)(ws + 256 + 2 * szX + 4 * szW);
  f16* Vh  = (f16*)(ws + 256 + 3 * szX + 4 * szW);
  f16* Ch  = (f16*)(ws + 256 + 4 * szX + 4 * szW);

  // 1) routing
  route_kernel<<<Bz, 256, 0, stream>>>(hidden, centers, assign);

  // 2) casts
  {
    int nX = Bz * Sq * Dm;
    cast_f16_kernel<<<(nX + 255) / 256, 256, 0, stream>>>(hidden, Xh, nX);
    int nW = 2 * Dm * Dm;
    cast_f16_kernel<<<(nW + 255) / 256, 256, 0, stream>>>(Wq, Wqh, nW);
    cast_f16_kernel<<<(nW + 255) / 256, 256, 0, stream>>>(Wk, Wkh, nW);
    cast_f16_kernel<<<(nW + 255) / 256, 256, 0, stream>>>(Wv, Wvh, nW);
    cast_f16_kernel<<<(nW + 255) / 256, 256, 0, stream>>>(Wo, Woh, nW);
  }

  // 3) Q/K/V projections (expert-routed)
  dim3 gg(Sq / 128, Dm / 128, Bz);
  gemm_bias_kernel<true><<<gg, 256, 0, stream>>>(Xh, Wqh, bq, assign, Qh, nullptr);
  gemm_bias_kernel<true><<<gg, 256, 0, stream>>>(Xh, Wkh, bk, assign, Kh, nullptr);
  gemm_bias_kernel<true><<<gg, 256, 0, stream>>>(Xh, Wvh, bv, assign, Vh, nullptr);

  // 4) attention
  attn_kernel<<<dim3(Bz * Hh, Sq / 32), 256, 0, stream>>>(Qh, Kh, Vh, amask, Ch);

  // 5) output projection -> f32 out
  gemm_bias_kernel<false><<<gg, 256, 0, stream>>>(Ch, Woh, bo, assign, nullptr, out);
}
